// QuantLinear_21861383537365
// MI455X (gfx1250) — matrix operand order repaired and checked by reading, not compiled, not run on hardware
//
#include <hip/hip_runtime.h>
#include <stdint.h>

#define TOKENS 16384
#define D_IN   2048
#define D_OUT  2048
#define BM 128
#define BN 128
#define BK 32
#define LDAF 36          // A-tile row stride in floats (144B, 16B-aligned)
#define LDB  40          // B-tile (transposed) row stride in halves (80B)
#define KTILES (D_IN / BK)

typedef _Float16 v16h __attribute__((ext_vector_type(16)));
typedef _Float16 v8h  __attribute__((ext_vector_type(8)));
typedef _Float16 v2h  __attribute__((ext_vector_type(2)));
typedef float    v8f  __attribute__((ext_vector_type(8)));
typedef float    v4f  __attribute__((ext_vector_type(4)));

// CDNA5 async direct-to-LDS copy (ASYNCcnt-tracked, no VGPR round trip).
__device__ __forceinline__ void async_copy_b128(uint32_t lds_off, const void* g) {
  asm volatile("global_load_async_to_lds_b128 %0, %1, off"
               :: "v"(lds_off), "v"((unsigned long long)(uintptr_t)g)
               : "memory");
}
__device__ __forceinline__ void wait_async_le4() {   // keep newest tile in flight
  asm volatile("s_wait_asynccnt 0x4" ::: "memory");
}
__device__ __forceinline__ void wait_async0() {
  asm volatile("s_wait_asynccnt 0x0" ::: "memory");
}

// magic-bias dequant: dword holds two f16 patterns 0x6400|u (= 1024+u);
// one packed subtract of 1152 yields the exact signed int8 value as f16.
__device__ __forceinline__ uint32_t dq2(uint32_t packed) {
  v2h v = __builtin_bit_cast(v2h, packed);
  const v2h magic = {(_Float16)1152, (_Float16)1152};
  v = v - magic;
  return __builtin_bit_cast(uint32_t, v);
}

__global__ __launch_bounds__(256)
void qlinear_wmma_f16(const float* __restrict__ x,
                      const float* __restrict__ scale,
                      const int8_t* __restrict__ w,
                      float* __restrict__ out) {
  __shared__ float    aLds[3][BM * LDAF];   // raw f32 x tiles (async, 3-deep)
  __shared__ _Float16 bLds[2][BN * LDB];    // dequantized W tile, [N][K]

  const int t     = threadIdx.x;
  const int lane  = t & 31;
  const int wave  = t >> 5;        // 0..7
  const int waveM = wave & 3;      // 0..3 -> 32 rows each
  const int waveN = wave >> 2;     // 0..1 -> 64 cols each
  const int n0    = blockIdx.x * BN;
  const int m0    = blockIdx.y * BM;

  const int  lhalf = lane & 15;
  const bool hi    = lane >= 16;

  // staging decomposition
  const int ar = t >> 3;           // A: base row 0..31 (strided by 32, x4)
  const int ac = (t & 7) * 4;      // A: col in floats (16B chunk)
  const int bg = t >> 5;           // B: k-group 0..7 -> rows 4bg..4bg+3
  const int bn = (t & 31) * 4;     // B: col base 0..124

  v8f acc[2][4] = {};              // 2 (M) x 4 (N) tiles of 16x16 f32

  // ---- A tile: 128x32 f32, four async 16B chunks per thread, direct to LDS.
  auto stageA = [&](int kt, int buf) {
    const int k0 = kt * BK;
#pragma unroll
    for (int j = 0; j < 4; ++j) {
      const int r = ar + 32 * j;
      const uint32_t loff = (uint32_t)(uintptr_t)&aLds[buf][r * LDAF + ac];
      async_copy_b128(loff, x + (size_t)(m0 + r) * D_IN + (k0 + ac));
    }
  };

  // ---- B tile: each thread owns a 4(K)x4(N) micro-tile of W.
  // 4 coalesced b32 loads; perm-network transpose + magic-bias dequant
  // (exact for int8); b64 stores into the [N][K] f16 layout.
  auto stageB = [&](int kt, int buf) {
    const int k0 = kt * BK;
    uint32_t r[4];
#pragma unroll
    for (int j = 0; j < 4; ++j) {
      const int8_t* bp = w + (size_t)(k0 + 4 * bg + j) * D_OUT + (n0 + bn);
      r[j] = *(const uint32_t*)bp ^ 0x80808080u;   // bias int8 -> [0,255]
    }
    __builtin_prefetch(w + (size_t)(k0 + 2 * BK + 4 * bg) * D_OUT + (n0 + bn),
                       0, 0);
    uint32_t q[4][2];                // [column n][k-pair p]
#pragma unroll
    for (int p = 0; p < 2; ++p) {    // k-pairs (4bg+2p, 4bg+2p+1)
      const uint32_t ra = r[2 * p], rb = r[2 * p + 1];
      // interleave bytes of the two k-rows: {rb1,ra1,rb0,ra0}, {rb3,ra3,rb2,ra2}
      const uint32_t tlo = __builtin_amdgcn_perm(rb, ra, 0x05010400u);
      const uint32_t thi = __builtin_amdgcn_perm(rb, ra, 0x07030602u);
      // expand to f16 bit patterns {0x64,u_k1, 0x64,u_k0} per column, dequant
      q[0][p] = dq2(__builtin_amdgcn_perm(0x64646464u, tlo, 0x05010400u));
      q[1][p] = dq2(__builtin_amdgcn_perm(0x64646464u, tlo, 0x05030402u));
      q[2][p] = dq2(__builtin_amdgcn_perm(0x64646464u, thi, 0x05010400u));
      q[3][p] = dq2(__builtin_amdgcn_perm(0x64646464u, thi, 0x05030402u));
    }
#pragma unroll
    for (int i = 0; i < 4; ++i)      // one b64 store per column: k=4bg..4bg+3
      *(uint2*)&bLds[buf][(bn + i) * LDB + 4 * bg] =
          make_uint2(q[i][0], q[i][1]);
  };

  // Prologue: A tiles 0 and 1 in flight, B tile 0 staged.
  stageA(0, 0);
  stageA(1, 1);
  stageB(0, 0);
  wait_async_le4();                // tile 0 landed, tile 1 may still fly
  __syncthreads();

  int aCur = 0;
  for (int kt = 0; kt < KTILES; ++kt) {
    const int bCur = kt & 1;
    const int aNxt2 = (aCur == 0) ? 2 : aCur - 1;   // (aCur + 2) % 3

    if (kt + 2 < KTILES)
      stageA(kt + 2, aNxt2);
    if (kt + 1 < KTILES)
      stageB(kt + 1, bCur ^ 1);

    // ---- A fragments: read raw f32 from LDS, convert to f16 at load.
    // ISA 16-bit A 16x32 layout: lane<16 -> K[0..7]|K[16..23],
    //                            lane>=16 -> K[8..15]|K[24..31]
    v16h afrag[2];
    const int akoff = hi ? 8 : 0;
#pragma unroll
    for (int sm = 0; sm < 2; ++sm) {
      const float* ap =
          &aLds[aCur][(waveM * 32 + sm * 16 + lhalf) * LDAF + akoff];
      v4f f0 = *(const v4f*)(ap + 0);
      v4f f1 = *(const v4f*)(ap + 4);
      v4f f2 = *(const v4f*)(ap + 16);
      v4f f3 = *(const v4f*)(ap + 20);
      v16h a;
#pragma unroll
      for (int i = 0; i < 4; ++i) {
        a[i]      = (_Float16)f0[i];
        a[4 + i]  = (_Float16)f1[i];
        a[8 + i]  = (_Float16)f2[i];
        a[12 + i] = (_Float16)f3[i];
      }
      afrag[sm] = a;
    }

    // ---- B fragments: lane<16 holds K0..7|K16..23, lane>=16 holds K8..15|K24..31
    // -> two 16-byte chunks in the transposed LDS tile.
    v16h bfrag[4];
    const int bkoff = hi ? 8 : 0;
#pragma unroll
    for (int sn = 0; sn < 4; ++sn) {
      const _Float16* bp =
          &bLds[bCur][(waveN * 64 + sn * 16 + lhalf) * LDB + bkoff];
      v8h lo = *(const v8h*)(bp);
      v8h hh = *(const v8h*)(bp + 16);
      bfrag[sn] = __builtin_shufflevector(lo, hh, 0, 1, 2, 3, 4, 5, 6, 7, 8, 9,
                                          10, 11, 12, 13, 14, 15);
    }

#pragma unroll
    for (int sm = 0; sm < 2; ++sm)
#pragma unroll
      for (int sn = 0; sn < 4; ++sn)
        acc[sm][sn] = __builtin_amdgcn_wmma_f32_16x16x32_f16(
            false, afrag[sm], false, bfrag[sn], (short)0, acc[sm][sn], false,
            false);

    // Ensure tile kt+1's async copies landed; tile kt+2's may stay in flight.
    if (kt + 2 < KTILES)
      wait_async_le4();
    else
      wait_async0();
    __syncthreads();

    aCur = (aCur == 2) ? 0 : aCur + 1;
  }

  // ---- Epilogue: per-output-channel dequant scale + store.
  // C/D layout: VGPR j -> M = j (lanes 0-15) / j+8 (lanes 16-31), N = lane&15.
#pragma unroll
  for (int sn = 0; sn < 4; ++sn) {
    const int col  = n0 + waveN * 64 + sn * 16 + lhalf;
    const float sc = scale[col];
#pragma unroll
    for (int sm = 0; sm < 2; ++sm) {
      const int rbase = m0 + waveM * 32 + sm * 16 + (hi ? 8 : 0);
#pragma unroll
      for (int j = 0; j < 8; ++j)
        out[(size_t)(rbase + j) * D_OUT + col] = acc[sm][sn][j] * sc;
    }
  }
}

extern "C" void kernel_launch(void* const* d_in, const int* in_sizes, int n_in,
                              void* d_out, int out_size, void* d_ws,
                              size_t ws_size, hipStream_t stream) {
  (void)in_sizes; (void)n_in; (void)out_size; (void)d_ws; (void)ws_size;
  const float*  x     = (const float*)d_in[0];
  const float*  scale = (const float*)d_in[1];
  const int8_t* w     = (const int8_t*)d_in[2];
  float*        out   = (float*)d_out;

  dim3 grid(D_OUT / BN, TOKENS / BM);   // (16, 128) blocks
  qlinear_wmma_f16<<<grid, 256, 0, stream>>>(x, scale, w, out);
}
